// ApproxAttention_27307402068271
// MI455X (gfx1250) — compile-verified, matrix operand order checked by reading, not run on hardware
//
#include <hip/hip_runtime.h>

typedef __attribute__((ext_vector_type(16))) _Float16 v16h;
typedef __attribute__((ext_vector_type(8)))  float    v8f;

#define B_  2
#define H_  16
#define S_  2048
#define D_  64
#define QT  16          // query rows per workgroup
#define NT  256         // threads per workgroup (8 waves)

// LDS layout offsets (bytes)
#define OFF_SS   0              // float [16][2048]  = 131072
#define OFF_SO   131072         // float [16][64]    = 4096
#define OFF_RED  135168         // float [16][16]    = 1024
#define OFF_MAX  136192         // float [16]        = 64
#define OFF_INV  136256         // float [16]        = 64
#define OFF_SQ   136320         // f16   [16][64]    = 2048
#define OFF_SKV  138368         // f16 staging       = 32768
#define SMEM_BYTES 171136

// K-index pattern for 16-bit A/B fragments (CDNA5 16x32 layout):
// VGPR v (0..7): base = (v>>2)*16, within-group = (v&3)*2, lane-half adds 8.
__device__ __forceinline__ int kpat(int v, int hlf) {
    return ((v >> 2) << 4) + hlf * 8 + ((v & 3) << 1);
}

__global__ __launch_bounds__(NT)
void attn_fused_kernel(const float* __restrict__ Qg,
                       const float* __restrict__ Kg,
                       const float* __restrict__ Vg,
                       float* __restrict__ outP,
                       float* __restrict__ pAttn)
{
    extern __shared__ char smem[];
    float*    sS   = (float*)(smem + OFF_SS);    // scores [16][2048]
    float*    sO   = (float*)(smem + OFF_SO);    // out accum [16][64]
    float*    sRed = (float*)(smem + OFF_RED);   // reduction scratch [16][16]
    float*    sMax = (float*)(smem + OFF_MAX);   // row max [16]
    float*    sInv = (float*)(smem + OFF_INV);   // 1/rowsum [16]
    _Float16* sQ   = (_Float16*)(smem + OFF_SQ); // Q tile f16 [16][64]
    _Float16* sKV  = (_Float16*)(smem + OFF_SKV);// per-wave K/V staging

    const int tid  = threadIdx.x;
    const int lane = tid & 31;
    const int wave = tid >> 5;
    const int sub  = lane & 15;        // N / M index within fragment
    const int hlf  = (lane >> 4) & 1;  // lane half

    const int gid = blockIdx.x;        // 0 .. B*H*(S/QT)-1
    const int bh  = gid >> 7;          // S/QT = 128 query tiles per (b,h)
    const int qt  = gid & 127;

    const size_t qkvBase = (size_t)bh * S_ * D_;
    const size_t qBase   = qkvBase + (size_t)qt * QT * D_;
    const size_t pBase   = (size_t)bh * S_ * S_ + (size_t)qt * QT * S_;

    // ---- stage Q (f32 -> f16), zero output accumulator ----
    for (int i = tid; i < QT * D_; i += NT) {
        sO[i] = 0.0f;
        sQ[i] = (_Float16)Qg[qBase + i];
    }
    __syncthreads();

    // ---- per-wave A fragments of Q: two K-chunks of 32 along D ----
    v16h aQ0, aQ1;
#pragma unroll
    for (int v = 0; v < 8; ++v) {
        const int k0 = kpat(v, hlf);
        aQ0[2 * v]     = sQ[sub * D_ + k0];
        aQ0[2 * v + 1] = sQ[sub * D_ + k0 + 1];
        aQ1[2 * v]     = sQ[sub * D_ + 32 + k0];
        aQ1[2 * v + 1] = sQ[sub * D_ + 32 + k0 + 1];
    }

    // ================= Phase 1: scores = (Q K^T) * 1/sqrt(D) =================
    {
        _Float16* sKw = sKV + wave * (16 * D_);
        const float scale = 0.125f; // 1/sqrt(64)
        for (int t = wave * 16; t < wave * 16 + 16; ++t) {
            const int keyBase = t * 16;
            // stage K tile 16x64 (coalesced f32 loads, convert to f16)
            for (int j = lane; j < 16 * D_; j += 32)
                sKw[j] = (_Float16)Kg[qkvBase + (size_t)keyBase * D_ + j];
            if (t + 1 < wave * 16 + 16)
                __builtin_prefetch(&Kg[qkvBase + (size_t)(keyBase + 16) * D_ + lane * 8], 0, 1);

            v8f c = {};
            v16h b0, b1;
#pragma unroll
            for (int v = 0; v < 8; ++v) {
                const int k0 = kpat(v, hlf);
                // B = K^T : element (k=d, n=key) = K[key][d]; key = sub
                b0[2 * v]     = sKw[sub * D_ + k0];
                b0[2 * v + 1] = sKw[sub * D_ + k0 + 1];
                b1[2 * v]     = sKw[sub * D_ + 32 + k0];
                b1[2 * v + 1] = sKw[sub * D_ + 32 + k0 + 1];
            }
            c = __builtin_amdgcn_wmma_f32_16x16x32_f16(false, aQ0, false, b0,
                                                       (short)0, c, false, false);
            c = __builtin_amdgcn_wmma_f32_16x16x32_f16(false, aQ1, false, b1,
                                                       (short)0, c, false, false);
            // D layout: VGPR v -> row (v + hlf*8), col = sub
#pragma unroll
            for (int v = 0; v < 8; ++v)
                sS[(v + hlf * 8) * S_ + keyBase + sub] = c[v] * scale;
        }
    }
    __syncthreads();

    // ================= Phase 2: softmax over rows of sS, write p_attn ========
    {
        const int r = tid >> 4;   // row 0..15
        const int c = tid & 15;   // sub-column 0..15 (stride-16 lane pattern)
        float m = -3.0e38f;
        for (int j = 0; j < S_ / 16; ++j)
            m = fmaxf(m, sS[r * S_ + j * 16 + c]);
        sRed[r * 16 + c] = m;
        __syncthreads();
        if (tid < 16) {
            float mm = sRed[tid * 16];
            for (int j = 1; j < 16; ++j) mm = fmaxf(mm, sRed[tid * 16 + j]);
            sMax[tid] = mm;
        }
        __syncthreads();
        const float rm = sMax[r];
        float acc = 0.0f;
        for (int j = 0; j < S_ / 16; ++j) {
            const int idx = r * S_ + j * 16 + c;
            const float e = __expf(sS[idx] - rm);
            sS[idx] = e;
            acc += e;
        }
        sRed[r * 16 + c] = acc;
        __syncthreads();
        if (tid < 16) {
            float s = 0.0f;
            for (int j = 0; j < 16; ++j) s += sRed[tid * 16 + j];
            sInv[tid] = 1.0f / s;
        }
        __syncthreads();
        // normalize in LDS and stream p_attn to global (coalesced)
        for (int i = tid; i < QT * S_; i += NT) {
            const int rr = i >> 11;
            const float p = sS[i] * sInv[rr];
            sS[i] = p;
            pAttn[pBase + i] = p;
        }
    }
    __syncthreads();

    // ================= Phase 3: out = P @ V ==================================
    {
        v8f acc[4] = {v8f{}, v8f{}, v8f{}, v8f{}};
        _Float16* sVw = sKV + wave * (32 * D_);
        for (int ch = wave * 8; ch < wave * 8 + 8; ++ch) {
            const int keyBase = ch * 32;
            // stage V tile 32x64 (coalesced, f32 -> f16)
            for (int j = lane; j < 32 * D_; j += 32)
                sVw[j] = (_Float16)Vg[qkvBase + (size_t)keyBase * D_ + j];

            // A = P[16 rows][32 keys] from normalized LDS scores
            v16h a;
#pragma unroll
            for (int v = 0; v < 8; ++v) {
                const int k0 = kpat(v, hlf);
                a[2 * v]     = (_Float16)sS[sub * S_ + keyBase + k0];
                a[2 * v + 1] = (_Float16)sS[sub * S_ + keyBase + k0 + 1];
            }
#pragma unroll
            for (int dt = 0; dt < 4; ++dt) {
                // B = V[32 keys][16 d-cols]; element (k=key, n=d) = V[key][d]
                v16h b;
#pragma unroll
                for (int v = 0; v < 8; ++v) {
                    const int k0 = kpat(v, hlf);
                    b[2 * v]     = sVw[k0 * D_ + dt * 16 + sub];
                    b[2 * v + 1] = sVw[(k0 + 1) * D_ + dt * 16 + sub];
                }
                acc[dt] = __builtin_amdgcn_wmma_f32_16x16x32_f16(false, a, false, b,
                                                                 (short)0, acc[dt],
                                                                 false, false);
            }
        }
        // cross-wave reduction of partial outputs via LDS float atomics
#pragma unroll
        for (int dt = 0; dt < 4; ++dt)
#pragma unroll
            for (int v = 0; v < 8; ++v)
                atomicAdd(&sO[(v + hlf * 8) * D_ + dt * 16 + sub], acc[dt][v]);
    }
    __syncthreads();

    for (int i = tid; i < QT * D_; i += NT)
        outP[qBase + i] = sO[i];
}

extern "C" void kernel_launch(void* const* d_in, const int* in_sizes, int n_in,
                              void* d_out, int out_size, void* d_ws, size_t ws_size,
                              hipStream_t stream) {
    (void)in_sizes; (void)n_in; (void)out_size; (void)d_ws; (void)ws_size;
    const float* Q = (const float*)d_in[0];
    const float* K = (const float*)d_in[1];
    const float* V = (const float*)d_in[2];
    float* out   = (float*)d_out;
    float* pattn = out + (size_t)B_ * H_ * S_ * D_;

    dim3 grid(B_ * H_ * (S_ / QT));   // 4096 workgroups
    dim3 block(NT);
    hipLaunchKernelGGL(attn_fused_kernel, grid, block, SMEM_BYTES, stream,
                       Q, K, V, out, pattn);
}
